// TransformerLayer_6511170421097
// MI455X (gfx1250) — matrix operand order repaired and checked by reading, not compiled, not run on hardware
//
#include <hip/hip_runtime.h>

typedef unsigned short u16;
typedef __bf16 bf16_t;
typedef bf16_t v16bf __attribute__((ext_vector_type(16)));
typedef float  v8f   __attribute__((ext_vector_type(8)));

union Frag {
    uint4 u[2];
    v16bf v;
};

__device__ __forceinline__ u16 f2bf(float x) {
    unsigned int u = __float_as_uint(x);
    unsigned int r = u + 0x7FFFu + ((u >> 16) & 1u);
    return (u16)(r >> 16);
}

__device__ __forceinline__ float gelu_tanh(float x) {
    float x3 = x * x * x;
    float t = tanhf(0.7978845608028654f * (x + 0.044715f * x3));
    return 0.5f * x * (1.0f + t);
}

// ---------------------------------------------------------------------------
// Transpose fp32 [K x N] -> bf16 [N x K]
// ---------------------------------------------------------------------------
__global__ __launch_bounds__(256) void transpose_bf16_kernel(
    const float* __restrict__ W, u16* __restrict__ WT, int K, int N)
{
    __shared__ u16 tile[32][33];
    int tx = threadIdx.x, ty = threadIdx.y;      // block (32, 8)
    int n0 = blockIdx.x * 32, k0 = blockIdx.y * 32;
#pragma unroll
    for (int i = 0; i < 4; i++) {
        int kk = ty + i * 8;
        tile[kk][tx] = f2bf(W[(size_t)(k0 + kk) * N + (n0 + tx)]);
    }
    __syncthreads();
#pragma unroll
    for (int i = 0; i < 4; i++) {
        int nn = ty + i * 8;
        WT[(size_t)(n0 + nn) * K + (k0 + tx)] = tile[tx][nn];
    }
}

// ---------------------------------------------------------------------------
// LayerNorm over D=2048, one row per block (256 threads).
// y = ln(in)*g + b ; optional y += res ; writes f32 and/or bf16.
// ---------------------------------------------------------------------------
__global__ __launch_bounds__(256) void ln_kernel(
    const float* __restrict__ in, const float* __restrict__ g,
    const float* __restrict__ be, const float* __restrict__ res,
    float* __restrict__ outf, u16* __restrict__ outb)
{
    const int D = 2048;
    __shared__ float rs[256], rq[256];
    int row = blockIdx.x, tid = threadIdx.x;
    const float* p = in + (size_t)row * D;
    float v[8];
    float s = 0.f, q = 0.f;
#pragma unroll
    for (int i = 0; i < 8; i++) {
        v[i] = p[tid + i * 256];
        s += v[i];
        q += v[i] * v[i];
    }
    rs[tid] = s; rq[tid] = q;
    __syncthreads();
    for (int off = 128; off > 0; off >>= 1) {
        if (tid < off) { rs[tid] += rs[tid + off]; rq[tid] += rq[tid + off]; }
        __syncthreads();
    }
    float mean = rs[0] * (1.0f / D);
    float var  = rq[0] * (1.0f / D) - mean * mean;
    float rstd = rsqrtf(var + 1e-5f);
#pragma unroll
    for (int i = 0; i < 8; i++) {
        int c = tid + i * 256;
        float y = (v[i] - mean) * rstd * g[c] + be[c];
        if (res)  y += res[(size_t)row * D + c];
        if (outf) outf[(size_t)row * D + c] = y;
        if (outb) outb[(size_t)row * D + c] = f2bf(y);
    }
}

// ---------------------------------------------------------------------------
// C[M,N] = A[MxK](bf16) * BT[NxK](bf16)^T + bias, optional gelu.
// 128x128 block tile, 8 waves (2x4), each wave 64x32 via 4x2 wmma tiles.
// ---------------------------------------------------------------------------
__global__ __launch_bounds__(256) void gemm_nt_kernel(
    const u16* __restrict__ A, const u16* __restrict__ BT,
    const float* __restrict__ bias, float* __restrict__ Cf,
    u16* __restrict__ Cb, int M, int N, int K, int dogelu)
{
    __shared__ u16 As[128 * 32];
    __shared__ u16 Bs[128 * 32];
    const int tid = threadIdx.x;
    const int wave = tid >> 5, lane = tid & 31;
    const int l16 = lane & 15, hi = lane >> 4;
    const int wm = wave >> 2, wn = wave & 3;
    const int mBlk = blockIdx.y * 128;
    const int nBlk = blockIdx.x * 128;

    v8f acc[4][2];
#pragma unroll
    for (int t = 0; t < 4; t++)
#pragma unroll
        for (int u = 0; u < 2; u++)
            acc[t][u] = (v8f){0.f, 0.f, 0.f, 0.f, 0.f, 0.f, 0.f, 0.f};

    const int abase = hi ? 8 : 0;   // A-frag K chunk base (elements)
    const int bbase = hi ? 8 : 0;   // B-frag K base (elements)

    for (int k0 = 0; k0 < K; k0 += 32) {
        __syncthreads();
#pragma unroll
        for (int i = 0; i < 2; i++) {
            int qc = tid + i * 256;          // 512 chunks of 8 elems
            int row = qc >> 2;
            int c = (qc & 3) * 8;
            *(uint4*)(&As[row * 32 + c]) =
                *(const uint4*)(A + (size_t)(mBlk + row) * K + k0 + c);
            *(uint4*)(&Bs[row * 32 + c]) =
                *(const uint4*)(BT + (size_t)(nBlk + row) * K + k0 + c);
        }
        __syncthreads();

        Frag a[4], b[2];
#pragma unroll
        for (int t = 0; t < 4; t++) {
            int r = wm * 64 + t * 16 + l16;
            a[t].u[0] = *(const uint4*)(&As[r * 32 + abase]);
            a[t].u[1] = *(const uint4*)(&As[r * 32 + 16 + abase]);
        }
#pragma unroll
        for (int u = 0; u < 2; u++) {
            int r = wn * 32 + u * 16 + l16;
            b[u].u[0] = *(const uint4*)(&Bs[r * 32 + bbase]);
            b[u].u[1] = *(const uint4*)(&Bs[r * 32 + bbase + 16]);
        }
#pragma unroll
        for (int t = 0; t < 4; t++)
#pragma unroll
            for (int u = 0; u < 2; u++)
                acc[t][u] = __builtin_amdgcn_wmma_f32_16x16x32_bf16(
                    false, a[t].v, false, b[u].v, (short)0, acc[t][u], false, false);
    }

#pragma unroll
    for (int t = 0; t < 4; t++)
#pragma unroll
        for (int u = 0; u < 2; u++) {
            int n = nBlk + wn * 32 + u * 16 + l16;
            float bv = bias ? bias[n] : 0.f;
#pragma unroll
            for (int r = 0; r < 8; r++) {
                int m = mBlk + wm * 64 + t * 16 + r + hi * 8;
                float val = acc[t][u][r] + bv;
                if (dogelu) val = gelu_tanh(val);
                if (Cf) Cf[(size_t)m * N + n] = val;
                if (Cb) Cb[(size_t)m * N + n] = f2bf(val);
            }
        }
}

// ---------------------------------------------------------------------------
// Flash attention: causal, S=2048, H=16, DH=128, qkv bf16 [S, 3*2048].
// Block = 128 queries x 1 head, 8 waves x 16 queries, KV blocks of 32.
// ctx out: bf16 [S, 2048] (col = h*128 + dh).
// ---------------------------------------------------------------------------
__global__ __launch_bounds__(256) void flash_attn_kernel(
    const u16* __restrict__ qkv, u16* __restrict__ ctx)
{
    const int TD = 6144, DH = 128, D = 2048;
    __shared__ u16 VT[128 * 32];       // [dh][key] transposed V block
    __shared__ u16 Ps[8 * 16 * 32];    // per-wave P tile 16x32

    const int tid = threadIdx.x;
    const int wave = tid >> 5, lane = tid & 31;
    const int l16 = lane & 15, hi = lane >> 4;
    const int h = blockIdx.y;
    const int q0 = blockIdx.x * 128;
    const int qt = q0 + wave * 16;
    const int qcol = h * DH, kcol = D + h * DH, vcol = 2 * D + h * DH;
    const int abase = hi ? 8 : 0;
    const int bbase = hi ? 8 : 0;
    const float inv_scale = 0.08838834764831845f;  // 1/sqrt(128)

    // Preload Q fragments (4 K-steps of 32 over DH=128)
    Frag qf[4];
    {
        const u16* qrow = qkv + (size_t)(qt + l16) * TD + qcol;
#pragma unroll
        for (int f = 0; f < 4; f++) {
            qf[f].u[0] = *(const uint4*)(qrow + f * 32 + abase);
            qf[f].u[1] = *(const uint4*)(qrow + f * 32 + 16 + abase);
        }
    }

    v8f o[8];
#pragma unroll
    for (int u = 0; u < 8; u++) o[u] = (v8f){0.f,0.f,0.f,0.f,0.f,0.f,0.f,0.f};
    float mrow[8], lrow[8];
#pragma unroll
    for (int r = 0; r < 8; r++) { mrow[r] = -1e30f; lrow[r] = 0.f; }

    const int jmax = q0 / 32 + 4;      // covers keys <= q0+127
    for (int j = 0; j < jmax; j++) {
        const int kb = j * 32;
        __syncthreads();
        {   // stage V block transposed into LDS
            int r = tid >> 3;            // key row 0..31
            int c = (tid & 7) * 16;      // dh col base
            const u16* src = qkv + (size_t)(kb + r) * TD + vcol + c;
            u16 tmp[16];
            *(uint4*)(tmp)     = *(const uint4*)(src);
            *(uint4*)(tmp + 8) = *(const uint4*)(src + 8);
#pragma unroll
            for (int i = 0; i < 16; i++) VT[(c + i) * 32 + r] = tmp[i];
        }
        __syncthreads();

        // scores: S tile 16x32 = two 16x16 accumulators
        v8f s0 = (v8f){0.f,0.f,0.f,0.f,0.f,0.f,0.f,0.f};
        v8f s1 = s0;
#pragma unroll
        for (int f = 0; f < 4; f++) {
            Frag k0f, k1f;
            const u16* kr0 = qkv + (size_t)(kb + l16) * TD + kcol + f * 32 + bbase;
            const u16* kr1 = qkv + (size_t)(kb + 16 + l16) * TD + kcol + f * 32 + bbase;
            k0f.u[0] = *(const uint4*)(kr0); k0f.u[1] = *(const uint4*)(kr0 + 16);
            k1f.u[0] = *(const uint4*)(kr1); k1f.u[1] = *(const uint4*)(kr1 + 16);
            s0 = __builtin_amdgcn_wmma_f32_16x16x32_bf16(false, qf[f].v, false, k0f.v, (short)0, s0, false, false);
            s1 = __builtin_amdgcn_wmma_f32_16x16x32_bf16(false, qf[f].v, false, k1f.v, (short)0, s1, false, false);
        }

        // online softmax update per row (rows live across 16-lane halves)
        float scl[8];
#pragma unroll
        for (int r = 0; r < 8; r++) {
            int qg = qt + r + hi * 8;
            int key0 = kb + l16, key1 = key0 + 16;
            float v0 = s0[r] * inv_scale; if (key0 > qg) v0 = -10000.f;
            float v1 = s1[r] * inv_scale; if (key1 > qg) v1 = -10000.f;
            float t = fmaxf(v0, v1);
#pragma unroll
            for (int off = 1; off < 16; off <<= 1) t = fmaxf(t, __shfl_xor(t, off));
            float mn = fmaxf(mrow[r], t);
            float sc = expf(mrow[r] - mn);
            float p0 = expf(v0 - mn);
            float p1 = expf(v1 - mn);
            float rsum = p0 + p1;
#pragma unroll
            for (int off = 1; off < 16; off <<= 1) rsum += __shfl_xor(rsum, off);
            lrow[r] = lrow[r] * sc + rsum;
            mrow[r] = mn;
            scl[r] = sc;
            int Mloc = r + hi * 8;
            Ps[wave * 512 + Mloc * 32 + l16]      = f2bf(p0);
            Ps[wave * 512 + Mloc * 32 + 16 + l16] = f2bf(p1);
        }
#pragma unroll
        for (int u = 0; u < 8; u++)
#pragma unroll
            for (int r = 0; r < 8; r++) o[u][r] *= scl[r];

        __syncthreads();   // P visible to all lanes of wave, VT stable

        // P(16x32) @ V(32x128): A-frag from LDS, B-frags from VT
        Frag pf;
        pf.u[0] = *(const uint4*)(&Ps[wave * 512 + l16 * 32 + abase]);
        pf.u[1] = *(const uint4*)(&Ps[wave * 512 + l16 * 32 + 16 + abase]);
#pragma unroll
        for (int u = 0; u < 8; u++) {
            Frag vf;
            const u16* vr = &VT[(u * 16 + l16) * 32 + bbase];
            vf.u[0] = *(const uint4*)(vr);
            vf.u[1] = *(const uint4*)(vr + 16);
            o[u] = __builtin_amdgcn_wmma_f32_16x16x32_bf16(false, pf.v, false, vf.v, (short)0, o[u], false, false);
        }
    }

    // finalize: ctx = o / l, store bf16
    float invl[8];
#pragma unroll
    for (int r = 0; r < 8; r++) invl[r] = 1.0f / lrow[r];
#pragma unroll
    for (int u = 0; u < 8; u++)
#pragma unroll
        for (int r = 0; r < 8; r++) {
            int m = qt + r + hi * 8;
            int c = h * DH + u * 16 + l16;
            ctx[(size_t)m * D + c] = f2bf(o[u][r] * invl[r]);
        }
}

// ---------------------------------------------------------------------------
// Host orchestration
// ---------------------------------------------------------------------------
extern "C" void kernel_launch(void* const* d_in, const int* in_sizes, int n_in,
                              void* d_out, int out_size, void* d_ws, size_t ws_size,
                              hipStream_t stream)
{
    (void)in_sizes; (void)n_in; (void)out_size;
    const float* x       = (const float*)d_in[0];
    const float* Wqkv    = (const float*)d_in[2];
    const float* bqkv    = (const float*)d_in[3];
    const float* Wo      = (const float*)d_in[4];
    const float* bo      = (const float*)d_in[5];
    const float* W1      = (const float*)d_in[6];
    const float* b1      = (const float*)d_in[7];
    const float* W2      = (const float*)d_in[8];
    const float* b2      = (const float*)d_in[9];
    const float* g_ln_in = (const float*)d_in[10];
    const float* b_ln_in = (const float*)d_in[11];
    const float* g_s1    = (const float*)d_in[12];
    const float* b_s1    = (const float*)d_in[13];
    const float* g_ln_out= (const float*)d_in[14];
    const float* b_ln_out= (const float*)d_in[15];
    const float* g_s2    = (const float*)d_in[16];
    const float* b_s2    = (const float*)d_in[17];
    float* out = (float*)d_out;

    const int S = 2048, D = 2048;
    char* ws = (char*)d_ws;
    size_t used = 0;
    auto take = [&](size_t bytes) -> char* {
        char* p = ws + used;
        used += (bytes + 255) & ~(size_t)255;
        return p;
    };
    u16*   WqkvT = (u16*)  take((size_t)6144 * 2048 * 2);
    u16*   WoT   = (u16*)  take((size_t)2048 * 2048 * 2);
    u16*   W1T   = (u16*)  take((size_t)8192 * 2048 * 2);
    u16*   W2T   = (u16*)  take((size_t)2048 * 8192 * 2);
    u16*   h_b   = (u16*)  take((size_t)S * D * 2);
    u16*   qkv_b = (u16*)  take((size_t)S * 3 * D * 2);
    u16*   ctx_b = (u16*)  take((size_t)S * D * 2);
    float* attn_f= (float*)take((size_t)S * D * 4);
    float* x1_f  = (float*)take((size_t)S * D * 4);
    u16*   ho_b  = (u16*)  take((size_t)S * D * 2);
    u16*   u_b   = (u16*)  take((size_t)S * 4 * D * 2);
    float* mlp_f = (float*)take((size_t)S * D * 4);
    if (used > ws_size) return;

    dim3 tb(32, 8);
    // weight transpose+convert: W[K x N] -> WT[N x K] bf16
    transpose_bf16_kernel<<<dim3(6144 / 32, 2048 / 32), tb, 0, stream>>>(Wqkv, WqkvT, 2048, 6144);
    transpose_bf16_kernel<<<dim3(2048 / 32, 2048 / 32), tb, 0, stream>>>(Wo,   WoT,   2048, 2048);
    transpose_bf16_kernel<<<dim3(8192 / 32, 2048 / 32), tb, 0, stream>>>(W1,   W1T,   2048, 8192);
    transpose_bf16_kernel<<<dim3(2048 / 32, 8192 / 32), tb, 0, stream>>>(W2,   W2T,   8192, 2048);

    // h = ln(x) -> bf16
    ln_kernel<<<S, 256, 0, stream>>>(x, g_ln_in, b_ln_in, nullptr, nullptr, h_b);
    // qkv = h @ Wqkv + bqkv -> bf16
    gemm_nt_kernel<<<dim3(6144 / 128, S / 128), 256, 0, stream>>>(
        h_b, WqkvT, bqkv, nullptr, qkv_b, S, 6144, 2048, 0);
    // attention -> ctx bf16
    flash_attn_kernel<<<dim3(S / 128, 16), 256, 0, stream>>>(qkv_b, ctx_b);
    // attn = ctx @ Wo + bo -> f32
    gemm_nt_kernel<<<dim3(2048 / 128, S / 128), 256, 0, stream>>>(
        ctx_b, WoT, bo, attn_f, nullptr, S, 2048, 2048, 0);
    // x1 = x + ln(attn)
    ln_kernel<<<S, 256, 0, stream>>>(attn_f, g_s1, b_s1, x, x1_f, nullptr);
    // ho = ln(x1) -> bf16
    ln_kernel<<<S, 256, 0, stream>>>(x1_f, g_ln_out, b_ln_out, nullptr, nullptr, ho_b);
    // u = gelu(ho @ W1 + b1) -> bf16
    gemm_nt_kernel<<<dim3(8192 / 128, S / 128), 256, 0, stream>>>(
        ho_b, W1T, b1, nullptr, u_b, S, 8192, 2048, 1);
    // mlp = u @ W2 + b2 -> f32
    gemm_nt_kernel<<<dim3(2048 / 128, S / 128), 256, 0, stream>>>(
        u_b, W2T, b2, mlp_f, nullptr, S, 2048, 8192, 0);
    // out = x1 + ln(mlp)
    ln_kernel<<<S, 256, 0, stream>>>(mlp_f, g_s2, b_s2, x1_f, out, nullptr);
}
